// CoverageAttention_86998857548037
// MI455X (gfx1250) — compile-verified, matrix operand order checked by reading, not run on hardware
//
#include <hip/hip_runtime.h>
#include <hip/hip_bf16.h>
#include <math.h>

// Problem sizes
#define BB   32
#define LL   4096
#define ENC  512
#define DEC  512
#define ATT  512
#define COV  256
#define M_TOT (BB * LL)          // 131072 rows
#define NT   (ATT / 16)          // 32 N-tiles

typedef __attribute__((ext_vector_type(16))) __bf16 v16bf;
typedef __attribute__((ext_vector_type(8)))  float  v8f;
typedef __attribute__((ext_vector_type(4)))  unsigned int u32x4;
typedef __attribute__((ext_vector_type(8)))  int i32x8;
typedef __attribute__((ext_vector_type(4)))  int i32x4;

#if __has_builtin(__builtin_amdgcn_tensor_load_to_lds) && \
    __has_builtin(__builtin_amdgcn_s_wait_tensorcnt)
#define HAVE_TDM 1
#else
#define HAVE_TDM 0
#endif

// ---------- helpers ----------
static __device__ __forceinline__ unsigned short f2bf(float x) {
    unsigned int u = __float_as_uint(x);
    unsigned int r = u + 0x7FFFu + ((u >> 16) & 1u);   // round-to-nearest-even
    return (unsigned short)(r >> 16);
}

union FragU { int4 i[2]; v16bf v; };

static __device__ __forceinline__ v16bf ld_frag(const unsigned short* plo,
                                                const unsigned short* phi) {
    FragU u;
    u.i[0] = *reinterpret_cast<const int4*>(plo);
    u.i[1] = *reinterpret_cast<const int4*>(phi);
    return u.v;
}

#if HAVE_TDM
// 1-D TDM copy: ndw dwords from global -> LDS (D# per CDNA5 ISA ch.8:
// group0 = {count=1, lds_addr, global_addr, type=2}; group1 = {data_size=4B,
// tensor_dim0 = tile_dim0 = ndw}).  Issued by one wave; TENSORcnt tracked.
static __device__ __forceinline__ void tdm_copy_1d(unsigned lds_off,
                                                   const void* gptr,
                                                   unsigned ndw) {
    unsigned long long ga = (unsigned long long)gptr;
    u32x4 g0;
    g0[0] = 1u;                                     // count=1, user mode
    g0[1] = lds_off;                                // D#.lds_addr
    g0[2] = (unsigned)ga;                           // global_addr[31:0]
    g0[3] = (unsigned)(ga >> 32) | (2u << 30);      // global_addr[56:32] | type=2
    i32x8 g1;
    g1[0] = (int)(2u << 16);                        // wg_mask=0, data_size=4B
    g1[1] = (int)((ndw & 0xFFFFu) << 16);           // tensor_dim0[15:0]
    g1[2] = (int)((ndw >> 16) & 0xFFFFu);           // tensor_dim0[31:16]
    g1[3] = (int)((ndw & 0xFFFFu) << 16);           // tile_dim0 = ndw
    g1[4] = 0;                                      // tile_dim1/2 unused
    g1[5] = (int)ndw;                               // dim0 stride (benign)
    g1[6] = 0;
    g1[7] = 0;
    i32x4 z4 = {0, 0, 0, 0};
#if __clang_major__ >= 23
    i32x8 z8 = {0, 0, 0, 0, 0, 0, 0, 0};
    __builtin_amdgcn_tensor_load_to_lds(g0, g1, z4, z4, z8, 0);
#else
    __builtin_amdgcn_tensor_load_to_lds(g0, g1, z4, z4, 0);
#endif
}
#endif

// ---------- K1a: pack U_a / U_f to bf16 (same [a][k] layout == B-fragment [n][k]) ----------
__global__ void k_pack(const float* __restrict__ U_a, const float* __restrict__ U_f,
                       unsigned short* __restrict__ Uabf, unsigned short* __restrict__ Ufbf) {
    int idx = blockIdx.x * blockDim.x + threadIdx.x;
    if (idx < ATT * ENC) Uabf[idx] = f2bf(U_a[idx]);
    int j = idx - ATT * ENC;
    if (j >= 0 && j < ATT * COV) Ufbf[j] = f2bf(U_f[j]);
}

// ---------- K1b: Whp[b][a] = dec_h[b]·W_a[a] + b_Wa[a] + b_Ua[a] + b_Uf[a] ----------
__global__ void k_whp(const float* __restrict__ h, const float* __restrict__ W_a,
                      const float* __restrict__ b_Wa, const float* __restrict__ b_Ua,
                      const float* __restrict__ b_Uf, float* __restrict__ Whp) {
    int o = blockIdx.x * blockDim.x + threadIdx.x;      // 16384 outputs
    if (o >= BB * ATT) return;
    int b = o >> 9, a = o & (ATT - 1);
    float acc = b_Wa[a] + b_Ua[a] + b_Uf[a];
    const float* hr = h + b * DEC;
    const float* wr = W_a + a * DEC;
    #pragma unroll 8
    for (int d = 0; d < DEC; ++d) acc = fmaf(hr[d], wr[d], acc);
    Whp[o] = acc;
}

// ---------- K2: fused scores = v·tanh(Whp + enc@Uaᵀ + cov@Ufᵀ) + b_v  (WMMA bf16) ----------
// Each wave owns TWO 16-row M-tiles sharing one B fragment: 3 ds_load_b128 per WMMA.
#define WAVES_PER_WG 4
#define ROWS_PER_WG  (WAVES_PER_WG * 32)  // 128 rows
#define LDS_BA0   0                       // 16*512 ushort  = 16384 B (B slab buf 0)
#define LDS_BA1   16384                   // 16384 B (B slab buf 1)
#define LDS_BF0   32768                   //  8192 B
#define LDS_BF1   40960                   //  8192 B
#define LDS_AE    49152                   // 128*512 ushort = 131072 B
#define LDS_AC    180224                  // 128*256 ushort =  65536 B
#define LDS_TOTAL 245760

__global__ __launch_bounds__(128) void
k_scores(const float* __restrict__ enc, const float* __restrict__ cov,
         const unsigned short* __restrict__ Uabf, const unsigned short* __restrict__ Ufbf,
         const float* __restrict__ Whp, const float* __restrict__ v,
         const float* __restrict__ b_v, float* __restrict__ scores) {
    extern __shared__ char smem_dyn[];
    unsigned short* ldsAe = (unsigned short*)(smem_dyn + LDS_AE);
    unsigned short* ldsAc = (unsigned short*)(smem_dyn + LDS_AC);

    const int tid  = threadIdx.x;
    const int lane = tid & 31;
    const int wave = tid >> 5;
    const int m0_wg = blockIdx.x * ROWS_PER_WG;          // 128 rows per WG
    const int m0    = m0_wg + wave * 32;                 // wave: rows m0..m0+31
    const int batch = m0 >> 12;                          // 4096 % 128 == 0 -> uniform

    // ---- stage A tiles: fp32 -> bf16 into LDS (reused across all 32 N-tiles) ----
    for (int idx = tid; idx < 128 * (ENC / 4); idx += 128) {      // enc: 16384 float4
        int row = idx >> 7, c4 = idx & 127;
        const float4 x = *reinterpret_cast<const float4*>(
            enc + (size_t)(m0_wg + row) * ENC + c4 * 4);
        unsigned int p0 = ((unsigned int)f2bf(x.y) << 16) | f2bf(x.x);
        unsigned int p1 = ((unsigned int)f2bf(x.w) << 16) | f2bf(x.z);
        *reinterpret_cast<uint2*>(ldsAe + row * ENC + c4 * 4) = make_uint2(p0, p1);
    }
    for (int idx = tid; idx < 128 * (COV / 4); idx += 128) {      // cov: 8192 float4
        int row = idx >> 6, c4 = idx & 63;
        const float4 x = *reinterpret_cast<const float4*>(
            cov + (size_t)(m0_wg + row) * COV + c4 * 4);
        unsigned int p0 = ((unsigned int)f2bf(x.y) << 16) | f2bf(x.x);
        unsigned int p1 = ((unsigned int)f2bf(x.w) << 16) | f2bf(x.z);
        *reinterpret_cast<uint2*>(ldsAc + row * COV + c4 * 4) = make_uint2(p0, p1);
    }

#if HAVE_TDM
    const unsigned smbase = (unsigned)(unsigned long long)(void*)smem_dyn;
    if (wave == 0) {                      // prologue DMA: slabs for nt=0 -> buf 0
        tdm_copy_1d(smbase + LDS_BA0, Uabf, (16 * ENC) / 2);
        tdm_copy_1d(smbase + LDS_BF0, Ufbf, (16 * COV) / 2);
    }
#endif
    __syncthreads();

    const int half = lane >> 4;        // 0: K 0..7/16..23 ; 1: K 8..15/24..31
    const int mrow = lane & 15;        // A row index == B column index
    const unsigned short* Ae0 = ldsAe + (wave * 32 + mrow) * ENC;
    const unsigned short* Ae1 = Ae0 + 16 * ENC;
    const unsigned short* Ac0 = ldsAc + (wave * 32 + mrow) * COV;
    const unsigned short* Ac1 = Ac0 + 16 * COV;

    float rs0[8], rs1[8];
    #pragma unroll
    for (int r = 0; r < 8; ++r) { rs0[r] = 0.f; rs1[r] = 0.f; }

    for (int nt = 0; nt < NT; ++nt) {
        const int a0 = nt * 16;
        const unsigned short* bufBa;
        const unsigned short* bufBf;
#if HAVE_TDM
        // DMA next slabs into the other buffer while this one is consumed.
        if (wave == 0) {
            if (nt + 1 < NT) {
                const unsigned nb = (nt + 1) & 1;
                tdm_copy_1d(smbase + (nb ? LDS_BA1 : LDS_BA0),
                            Uabf + (nt + 1) * 16 * ENC, (16 * ENC) / 2);
                tdm_copy_1d(smbase + (nb ? LDS_BF1 : LDS_BF0),
                            Ufbf + (nt + 1) * 16 * COV, (16 * COV) / 2);
                __builtin_amdgcn_s_wait_tensorcnt(2);   // retire slabs for nt
            } else {
                __builtin_amdgcn_s_wait_tensorcnt(0);
            }
        }
        __syncthreads();
        bufBa = (const unsigned short*)(smem_dyn + ((nt & 1) ? LDS_BA1 : LDS_BA0));
        bufBf = (const unsigned short*)(smem_dyn + ((nt & 1) ? LDS_BF1 : LDS_BF0));
#else
        {   // fallback: cooperative copy of contiguous 16KB / 8KB slabs
            const int4* gBa = reinterpret_cast<const int4*>(Uabf + a0 * ENC);
            const int4* gBf = reinterpret_cast<const int4*>(Ufbf + a0 * COV);
            int4* dBa = reinterpret_cast<int4*>(smem_dyn + LDS_BA0);
            int4* dBf = reinterpret_cast<int4*>(smem_dyn + LDS_BF0);
            for (int i = tid; i < 1024; i += 128) dBa[i] = gBa[i];
            for (int i = tid; i < 512;  i += 128) dBf[i] = gBf[i];
        }
        __syncthreads();
        bufBa = (const unsigned short*)(smem_dyn + LDS_BA0);
        bufBf = (const unsigned short*)(smem_dyn + LDS_BF0);
#endif

        v8f acc0 = {}, acc1 = {};
        {   // enc @ Ua^T : software-pipelined, B fragment shared by two M-tiles
            const unsigned short* Bn = bufBa + mrow * ENC;
            v16bf a0A = ld_frag(Ae0 + half * 8, Ae0 + 16 + half * 8);
            v16bf a1A = ld_frag(Ae1 + half * 8, Ae1 + 16 + half * 8);
            v16bf bA  = ld_frag(Bn  + half * 8, Bn  + 16 + half * 8);
            #pragma unroll
            for (int kb = 0; kb < ENC; kb += 32) {
                v16bf a0B = a0A, a1B = a1A, bB = bA;
                if (kb + 32 < ENC) {
                    a0A = ld_frag(Ae0 + kb + 32 + half * 8, Ae0 + kb + 48 + half * 8);
                    a1A = ld_frag(Ae1 + kb + 32 + half * 8, Ae1 + kb + 48 + half * 8);
                    bA  = ld_frag(Bn  + kb + 32 + half * 8, Bn  + kb + 48 + half * 8);
                }
                acc0 = __builtin_amdgcn_wmma_f32_16x16x32_bf16(
                    false, a0B, false, bB, (short)0, acc0, false, false);
                acc1 = __builtin_amdgcn_wmma_f32_16x16x32_bf16(
                    false, a1B, false, bB, (short)0, acc1, false, false);
            }
        }
        {   // cov @ Uf^T
            const unsigned short* Bf = bufBf + mrow * COV;
            v16bf a0A = ld_frag(Ac0 + half * 8, Ac0 + 16 + half * 8);
            v16bf a1A = ld_frag(Ac1 + half * 8, Ac1 + 16 + half * 8);
            v16bf bA  = ld_frag(Bf  + half * 8, Bf  + 16 + half * 8);
            #pragma unroll
            for (int kb = 0; kb < COV; kb += 32) {
                v16bf a0B = a0A, a1B = a1A, bB = bA;
                if (kb + 32 < COV) {
                    a0A = ld_frag(Ac0 + kb + 32 + half * 8, Ac0 + kb + 48 + half * 8);
                    a1A = ld_frag(Ac1 + kb + 32 + half * 8, Ac1 + kb + 48 + half * 8);
                    bA  = ld_frag(Bf  + kb + 32 + half * 8, Bf  + kb + 48 + half * 8);
                }
                acc0 = __builtin_amdgcn_wmma_f32_16x16x32_bf16(
                    false, a0B, false, bB, (short)0, acc0, false, false);
                acc1 = __builtin_amdgcn_wmma_f32_16x16x32_bf16(
                    false, a1B, false, bB, (short)0, acc1, false, false);
            }
        }

        const float whp = Whp[batch * ATT + a0 + mrow];   // per-column bias (col = lane&15)
        const float vv  = v[a0 + mrow];
        #pragma unroll
        for (int r = 0; r < 8; ++r) {
            rs0[r] = fmaf(vv, tanhf(acc0[r] + whp), rs0[r]);
            rs1[r] = fmaf(vv, tanhf(acc1[r] + whp), rs1[r]);
        }
        __syncthreads();     // protect B buffer reuse (DMA/copy for nt+1 -> nt+2)
    }

    // reduce across the 16 lanes of each half-wave (xor masks stay inside the half)
    #pragma unroll
    for (int off = 1; off <= 8; off <<= 1)
        #pragma unroll
        for (int r = 0; r < 8; ++r) {
            rs0[r] += __shfl_xor(rs0[r], off, 32);
            rs1[r] += __shfl_xor(rs1[r], off, 32);
        }

    if (mrow == 0) {                   // lane 0 -> rows 0..7, lane 16 -> rows 8..15
        const float bv = b_v[0];
        #pragma unroll
        for (int r = 0; r < 8; ++r) {
            scores[m0 + half * 8 + r]      = rs0[r] + bv;
            scores[m0 + 16 + half * 8 + r] = rs1[r] + bv;
        }
    }
}

// ---------- K3a: per-batch softmax stats + zero the context output ----------
__global__ __launch_bounds__(256) void
k_stats(const float* __restrict__ scores, float* __restrict__ ctx_out,
        float* __restrict__ smax, float* __restrict__ sinv) {
    const int b = blockIdx.x, tid = threadIdx.x;
    ctx_out[b * ENC + tid] = 0.f;
    ctx_out[b * ENC + 256 + tid] = 0.f;

    __shared__ float red[256];
    float m = -INFINITY;
    for (int l = tid; l < LL; l += 256) m = fmaxf(m, scores[b * LL + l]);
    red[tid] = m; __syncthreads();
    for (int s = 128; s > 0; s >>= 1) {
        if (tid < s) red[tid] = fmaxf(red[tid], red[tid + s]);
        __syncthreads();
    }
    const float mx = red[0]; __syncthreads();

    float sum = 0.f;
    for (int l = tid; l < LL; l += 256) sum += __expf(scores[b * LL + l] - mx);
    red[tid] = sum; __syncthreads();
    for (int s = 128; s > 0; s >>= 1) {
        if (tid < s) red[tid] += red[tid + s];
        __syncthreads();
    }
    if (tid == 0) { smax[b] = mx; sinv[b] = 1.0f / red[0]; }
}

// ---------- K3b: alpha + context accumulation ----------
__global__ __launch_bounds__(256) void
k_ctx(const float* __restrict__ enc, const float* __restrict__ scores,
      const float* __restrict__ smax, const float* __restrict__ sinv,
      float* __restrict__ out) {
    const int b = blockIdx.x >> 4;
    const int l0 = (blockIdx.x & 15) * 256;
    const int tid = threadIdx.x;

    __shared__ float alpha_s[256];
    const float al = __expf(scores[b * LL + l0 + tid] - smax[b]) * sinv[b];
    alpha_s[tid] = al;
    out[BB * ENC + b * LL + l0 + tid] = al;              // alpha output slice
    __syncthreads();

    const int e0 = tid * 2;                              // 512 cols / 256 threads
    float ax = 0.f, ay = 0.f;
    for (int l = 0; l < 256; ++l) {
        const float2 e = *reinterpret_cast<const float2*>(
            enc + (size_t)(b * LL + l0 + l) * ENC + e0);
        const float a = alpha_s[l];
        ax = fmaf(a, e.x, ax);
        ay = fmaf(a, e.y, ay);
    }
    atomicAdd(&out[b * ENC + e0], ax);
    atomicAdd(&out[b * ENC + e0 + 1], ay);
}

// ---------- host ----------
extern "C" void kernel_launch(void* const* d_in, const int* in_sizes, int n_in,
                              void* d_out, int out_size, void* d_ws, size_t ws_size,
                              hipStream_t stream) {
    const float* enc  = (const float*)d_in[0];
    const float* hdec = (const float*)d_in[1];
    const float* cov  = (const float*)d_in[2];
    const float* W_a  = (const float*)d_in[3];
    const float* b_Wa = (const float*)d_in[4];
    const float* U_a  = (const float*)d_in[5];
    const float* b_Ua = (const float*)d_in[6];
    const float* U_f  = (const float*)d_in[7];
    const float* b_Uf = (const float*)d_in[8];
    const float* v    = (const float*)d_in[9];
    const float* b_v  = (const float*)d_in[10];
    float* out = (float*)d_out;

    char* w = (char*)d_ws;
    float*          Whp    = (float*)(w);                                   //  64 KB
    unsigned short* Uabf   = (unsigned short*)(w + 65536);                  // 512 KB
    unsigned short* Ufbf   = (unsigned short*)(w + 65536 + 524288);         // 256 KB
    float*          scores = (float*)(w + 65536 + 524288 + 262144);         // 512 KB
    float*          smax   = (float*)(w + 65536 + 524288 + 262144 + 524288);
    float*          sinv   = smax + 64;

    k_pack<<<(ATT * ENC + ATT * COV + 255) / 256, 256, 0, stream>>>(U_a, U_f, Uabf, Ufbf);
    k_whp<<<(BB * ATT + 255) / 256, 256, 0, stream>>>(hdec, W_a, b_Wa, b_Ua, b_Uf, Whp);
    k_scores<<<M_TOT / ROWS_PER_WG, 128, LDS_TOTAL, stream>>>(
        enc, cov, Uabf, Ufbf, Whp, v, b_v, scores);
    k_stats<<<BB, 256, 0, stream>>>(scores, out, smax, sinv);
    k_ctx<<<BB * 16, 256, 0, stream>>>(enc, scores, smax, sinv, out);
}
